// Attention_19928648254099
// MI455X (gfx1250) — matrix operand order repaired and checked by reading, not compiled, not run on hardware
//
#include <hip/hip_runtime.h>
#include <hip/hip_fp16.h>

// ---------------------------------------------------------------------------
// Attention: x[8,1024,768] -> qkv -> per-head softmax(QK^T/8)V -> proj + bias
// All GEMMs on v_wmma_f32_16x16x32_f16 (f32 accum), wave32.
// Weights pre-transposed at f16-convert time so every LDS stage is b128-copy;
// GEMM tiles double-buffered, staged via async-to-LDS (ASYNCcnt) when present.
// ---------------------------------------------------------------------------

typedef _Float16 f16_t;
typedef __attribute__((ext_vector_type(16))) _Float16 v16h;
typedef __attribute__((ext_vector_type(8)))  _Float16 v8h;
typedef __attribute__((ext_vector_type(8)))  float    v8f;

#define DIM   768
#define NH    12
#define HD    64
#define BATCH 8
#define SEQ   1024
#define MTOT  (BATCH * SEQ)          // 8192
#define SCALE_F 0.125f               // 64^-0.5

// --- opportunistic CDNA5 async global->LDS staging -------------------------
#ifndef USE_ASYNC_LDS
#  if defined(__has_builtin)
#    if __has_builtin(__builtin_amdgcn_global_load_async_to_lds_b128) && \
        __has_builtin(__builtin_amdgcn_s_wait_asynccnt)
#      define USE_ASYNC_LDS 1
#    else
#      define USE_ASYNC_LDS 0
#    endif
#  else
#    define USE_ASYNC_LDS 0
#  endif
#endif

#if USE_ASYNC_LDS
typedef int v4i_vs __attribute__((__vector_size__(16)));
typedef __attribute__((address_space(1))) v4i_vs* g_v4i_ptr;
typedef __attribute__((address_space(3))) v4i_vs* l_v4i_ptr;
#endif

__device__ __forceinline__ void copy_b128_to_lds(uint4* lds_dst,
                                                 const uint4* gsrc) {
#if USE_ASYNC_LDS
  __builtin_amdgcn_global_load_async_to_lds_b128(
      (g_v4i_ptr)(uint4*)gsrc, (l_v4i_ptr)lds_dst,
      /*imm offset*/ 0, /*cpol*/ 0);
#else
  *lds_dst = *gsrc;
#endif
}

__device__ __forceinline__ void async_stage_wait() {
#if USE_ASYNC_LDS
  __builtin_amdgcn_s_wait_asynccnt(0);
#endif
}

__device__ __forceinline__ v8f zero8() {
  v8f z;
#pragma unroll
  for (int i = 0; i < 8; ++i) z[i] = 0.0f;
  return z;
}

__device__ __forceinline__ v16h join8(v8h lo, v8h hi) {
  v16h r;
#pragma unroll
  for (int i = 0; i < 8; ++i) { r[i] = lo[i]; r[8 + i] = hi[i]; }
  return r;
}

// ---------------------------------------------------------------------------
// fp32 -> f16 convert (straight)
// ---------------------------------------------------------------------------
__global__ void cvt_f32_f16(const float* __restrict__ src,
                            f16_t* __restrict__ dst, int n) {
  int i = blockIdx.x * blockDim.x + threadIdx.x;
  int stride = gridDim.x * blockDim.x;
  for (; i < n; i += stride) dst[i] = (f16_t)src[i];
}

// fp32 [K][N] -> f16 [N][K]  (weight pre-transpose; reads coalesced)
__global__ void cvt_transpose_f32_f16(const float* __restrict__ src,
                                      f16_t* __restrict__ dst, int K, int N) {
  int i = blockIdx.x * blockDim.x + threadIdx.x;
  int stride = gridDim.x * blockDim.x;
  int total = K * N;
  for (; i < total; i += stride) {
    int k = i / N, n = i - k * N;
    dst[(size_t)n * K + k] = (f16_t)src[i];
  }
}

// ---------------------------------------------------------------------------
// Stage one 128x32-half tile (= 512 uint4) into LDS. Both A and B^T have
// K contiguous with row pitch 768 halfs (= 96 uint4). 256 threads, 2 each.
// ---------------------------------------------------------------------------
__device__ __forceinline__ void stage_tile(uint4* __restrict__ dst,
                                           const uint4* __restrict__ src,
                                           int row0, int col_u4) {
  const int tid = threadIdx.x;
#pragma unroll
  for (int it = 0; it < 2; ++it) {
    int i   = tid + it * 256;
    int row = i >> 2;              // 4 uint4 per 32-half row
    int q   = i & 3;
    copy_b128_to_lds(dst + i,
                     src + (size_t)(row0 + row) * (DIM / 8) + col_u4 + q);
  }
}

// ---------------------------------------------------------------------------
// 128x128 GEMM tile: C = A(128x768) * B^T(128x768)^T.  A row-major [M][K],
// BgT row-major [N][K] (pre-transposed weights).  256 threads = 8 waves,
// each wave a 64x32 sub-tile as 4x2 WMMA 16x16 accumulators.
// Double-buffered LDS; async staging overlaps the next tile with WMMA.
// ---------------------------------------------------------------------------
__device__ __forceinline__ void gemm_tile_128x128(
    const f16_t* __restrict__ Ag, const f16_t* __restrict__ BgT,
    int m0, int n0, uint4* Al, uint4* Bl, v8f acc[4][2]) {
  const int tid  = threadIdx.x;
  const int w    = tid >> 5;
  const int lane = tid & 31;
  const int l16  = lane & 15;
  const int hi   = lane >> 4;
  const int wm   = w >> 2;          // 0..1  (M)
  const int wn   = w & 3;           // 0..3  (N)
  const int aoff = hi ? 8 : 0;
  const int boff = hi ? 8 : 0;

#pragma unroll
  for (int mt = 0; mt < 4; ++mt)
#pragma unroll
    for (int nt = 0; nt < 2; ++nt) acc[mt][nt] = zero8();

  const uint4* A4 = (const uint4*)Ag;
  const uint4* B4 = (const uint4*)BgT;

  // prologue: stage k-chunk 0 into buffer 0
  stage_tile(Al, A4, m0, 0);
  stage_tile(Bl, B4, n0, 0);
  async_stage_wait();
  __syncthreads();

  for (int kb = 0; kb < DIM; kb += 32) {
    const int cur = (kb >> 5) & 1;
    if (kb + 32 < DIM) {           // stage next chunk into other buffer
      stage_tile(Al + (cur ^ 1) * 512, A4, m0, (kb + 32) >> 3);
      stage_tile(Bl + (cur ^ 1) * 512, B4, n0, (kb + 32) >> 3);
    }

    const f16_t* Af = (const f16_t*)(Al + cur * 512);
    const f16_t* Bf = (const f16_t*)(Bl + cur * 512);

    v16h af[4], bf[2];
#pragma unroll
    for (int mt = 0; mt < 4; ++mt) {
      const f16_t* ar = Af + (wm * 64 + mt * 16 + l16) * 32;
      v8h lo = *(const v8h*)(ar + aoff);
      v8h hh = *(const v8h*)(ar + aoff + 16);
      af[mt] = join8(lo, hh);
    }
#pragma unroll
    for (int nt = 0; nt < 2; ++nt)
      bf[nt] = join8(*(const v8h*)(Bf + (wn * 32 + nt * 16 + l16) * 32 + boff),
                     *(const v8h*)(Bf + (wn * 32 + nt * 16 + l16) * 32 + boff + 16));

#pragma unroll
    for (int mt = 0; mt < 4; ++mt)
#pragma unroll
      for (int nt = 0; nt < 2; ++nt)
        acc[mt][nt] = __builtin_amdgcn_wmma_f32_16x16x32_f16(
            false, af[mt], false, bf[nt], (short)0, acc[mt][nt], false, false);

    async_stage_wait();
    __syncthreads();
  }
}

// ---------------------------------------------------------------------------
// GEMM 1: qkv = X16 @ Wqkv.  Scatter into Q,K [B,H,N,D] (f16) and V
// transposed [B,H,D,N] (f16) so attention fragments are contiguous loads.
// grid (64, 18), block 256.
// ---------------------------------------------------------------------------
__global__ __launch_bounds__(256) void gemm_qkv_kernel(
    const f16_t* __restrict__ X, const f16_t* __restrict__ WT,
    f16_t* __restrict__ Q, f16_t* __restrict__ Ko, f16_t* __restrict__ Vt) {
  __shared__ __align__(32) uint4 Al[2 * 512];
  __shared__ __align__(32) uint4 Bl[2 * 512];
  v8f acc[4][2];
  const int m0 = blockIdx.x * 128;
  const int n0 = blockIdx.y * 128;
  gemm_tile_128x128(X, WT, m0, n0, Al, Bl, acc);

  const int tid = threadIdx.x;
  const int w = tid >> 5, lane = tid & 31, l16 = lane & 15, hi = lane >> 4;
  const int wm = w >> 2, wn = w & 3;
#pragma unroll
  for (int mt = 0; mt < 4; ++mt) {
#pragma unroll
    for (int nt = 0; nt < 2; ++nt) {
#pragma unroll
      for (int r = 0; r < 8; ++r) {
        int m = m0 + wm * 64 + mt * 16 + r + (hi ? 8 : 0);
        int c = n0 + wn * 32 + nt * 16 + l16;
        int which = c / DIM;        // 0=q 1=k 2=v (uniform per nt tile)
        int cc = c % DIM;
        int h = cc >> 6, d = cc & 63;
        int b = m >> 10, n = m & 1023;
        f16_t val = (f16_t)acc[mt][nt][r];
        size_t bh = (size_t)b * NH + h;
        if (which == 0)      Q [(bh * SEQ + n) * HD + d] = val;
        else if (which == 1) Ko[(bh * SEQ + n) * HD + d] = val;
        else                 Vt[(bh * HD + d) * SEQ + n] = val;
      }
    }
  }
}

// ---------------------------------------------------------------------------
// Flash attention: one block per (b, h, 64-row tile); 4 waves x 16 rows.
// S = Q K^T (WMMA), online softmax (shfl_xor row reductions), P staged via
// LDS into A-fragment layout, O += P V (WMMA).  Next K/V chunk prefetched.
// grid 1536, block 128.
// ---------------------------------------------------------------------------
__global__ __launch_bounds__(128) void attn_kernel(
    const f16_t* __restrict__ Q, const f16_t* __restrict__ Kb,
    const f16_t* __restrict__ Vt, f16_t* __restrict__ AO) {
  __shared__ __align__(32) f16_t Pl[4 * 16 * 64];   // per-wave 16x64 P tile
  const int tid  = threadIdx.x;
  const int w    = tid >> 5;
  const int lane = tid & 31;
  const int l16  = lane & 15;
  const int hi   = lane >> 4;
  const int bh   = blockIdx.x >> 4;                 // 0..95
  const int rt   = blockIdx.x & 15;
  const int m0   = rt * 64 + w * 16;
  const int b    = bh / NH, h = bh % NH;
  const int aoff = hi ? 8 : 0;
  const int boff = hi ? 8 : 0;

  // Q fragments for this wave's 16 rows (resident all kernel)
  const f16_t* qrow = Q + ((size_t)bh * SEQ + m0 + l16) * HD;
  v16h qa[2];
#pragma unroll
  for (int kb = 0; kb < 2; ++kb) {
    v8h lo = *(const v8h*)(qrow + kb * 32 + aoff);
    v8h hh = *(const v8h*)(qrow + kb * 32 + aoff + 16);
    qa[kb] = join8(lo, hh);
  }

  float mrow[8], lrow[8];
  v8f o[4];
#pragma unroll
  for (int r = 0; r < 8; ++r) { mrow[r] = -3.0e38f; lrow[r] = 0.0f; }
#pragma unroll
  for (int dt = 0; dt < 4; ++dt) o[dt] = zero8();

  f16_t* pw = Pl + w * (16 * 64);

  for (int nc = 0; nc < SEQ; nc += 64) {
    // prefetch next 64-column K/V chunk while this one computes
    if (nc + 64 < SEQ) {
      const f16_t* kn = Kb + ((size_t)bh * SEQ + nc + 64 + lane) * HD;
      __builtin_prefetch(kn, 0, 3);
      __builtin_prefetch(kn + 32 * HD, 0, 3);
      const f16_t* vn = Vt + ((size_t)bh * HD + lane) * SEQ + nc + 64;
      __builtin_prefetch(vn, 0, 3);
      __builtin_prefetch(vn + 32 * SEQ, 0, 3);
    }

    // ---- S = Q K^T over this 64-column chunk -------------------------------
    v8f s[4];
#pragma unroll
    for (int nt = 0; nt < 4; ++nt) s[nt] = zero8();
#pragma unroll
    for (int nt = 0; nt < 4; ++nt) {
      const f16_t* krow = Kb + ((size_t)bh * SEQ + nc + nt * 16 + l16) * HD;
#pragma unroll
      for (int kb = 0; kb < 2; ++kb) {
        v16h bf = join8(*(const v8h*)(krow + kb * 32 + boff),
                        *(const v8h*)(krow + kb * 32 + boff + 16));
        s[nt] = __builtin_amdgcn_wmma_f32_16x16x32_f16(
            false, qa[kb], false, bf, (short)0, s[nt], false, false);
      }
    }

    // ---- online softmax (rows r / r+8 in lanes 0-15 / 16-31) ---------------
    float p[4][8];
#pragma unroll
    for (int r = 0; r < 8; ++r) {
      float v0 = s[0][r] * SCALE_F, v1 = s[1][r] * SCALE_F;
      float v2 = s[2][r] * SCALE_F, v3 = s[3][r] * SCALE_F;
      float mx = fmaxf(fmaxf(v0, v1), fmaxf(v2, v3));
#pragma unroll
      for (int d = 1; d < 16; d <<= 1) mx = fmaxf(mx, __shfl_xor(mx, d, 32));
      float mnew = fmaxf(mrow[r], mx);
      float corr = __expf(mrow[r] - mnew);
      float p0 = __expf(v0 - mnew), p1 = __expf(v1 - mnew);
      float p2 = __expf(v2 - mnew), p3 = __expf(v3 - mnew);
      p[0][r] = p0; p[1][r] = p1; p[2][r] = p2; p[3][r] = p3;
      float rs = p0 + p1 + p2 + p3;
#pragma unroll
      for (int d = 1; d < 16; d <<= 1) rs += __shfl_xor(rs, d, 32);
      lrow[r] = lrow[r] * corr + rs;
      mrow[r] = mnew;
#pragma unroll
      for (int dt = 0; dt < 4; ++dt) o[dt][r] *= corr;
    }

    // ---- re-layout P (D-layout -> row-major 16x64) through LDS -------------
#pragma unroll
    for (int nt = 0; nt < 4; ++nt)
#pragma unroll
      for (int r = 0; r < 8; ++r)
        pw[(r + (hi ? 8 : 0)) * 64 + nt * 16 + l16] = (f16_t)p[nt][r];
    __syncthreads();

    v16h pa[2];
#pragma unroll
    for (int kb = 0; kb < 2; ++kb) {
      v8h lo = *(const v8h*)(pw + l16 * 64 + kb * 32 + aoff);
      v8h hh = *(const v8h*)(pw + l16 * 64 + kb * 32 + aoff + 16);
      pa[kb] = join8(lo, hh);
    }

    // ---- O += P V (V stored transposed -> contiguous B fragments) ----------
#pragma unroll
    for (int dt = 0; dt < 4; ++dt) {
      const f16_t* vcol = Vt + ((size_t)bh * HD + dt * 16 + l16) * SEQ + nc;
#pragma unroll
      for (int kb = 0; kb < 2; ++kb) {
        v16h bf = join8(*(const v8h*)(vcol + kb * 32 + boff),
                        *(const v8h*)(vcol + kb * 32 + boff + 16));
        o[dt] = __builtin_amdgcn_wmma_f32_16x16x32_f16(
            false, pa[kb], false, bf, (short)0, o[dt], false, false);
      }
    }
    __syncthreads();
  }

  // ---- normalize + store as [B, N, H*D] f16 for the proj GEMM --------------
  const size_t obase = (size_t)b * SEQ * DIM + (size_t)h * HD;
#pragma unroll
  for (int r = 0; r < 8; ++r) {
    float inv = 1.0f / lrow[r];
    int n = m0 + r + (hi ? 8 : 0);
#pragma unroll
    for (int dt = 0; dt < 4; ++dt)
      AO[obase + (size_t)n * DIM + dt * 16 + l16] = (f16_t)(o[dt][r] * inv);
  }
}

// ---------------------------------------------------------------------------
// GEMM 2: out = AO16 @ Wp + b_proj  (f32 output).  grid (64, 6), block 256.
// ---------------------------------------------------------------------------
__global__ __launch_bounds__(256) void gemm_proj_kernel(
    const f16_t* __restrict__ A, const f16_t* __restrict__ WT,
    const float* __restrict__ bias, float* __restrict__ Out) {
  __shared__ __align__(32) uint4 Al[2 * 512];
  __shared__ __align__(32) uint4 Bl[2 * 512];
  v8f acc[4][2];
  const int m0 = blockIdx.x * 128;
  const int n0 = blockIdx.y * 128;
  gemm_tile_128x128(A, WT, m0, n0, Al, Bl, acc);

  const int tid = threadIdx.x;
  const int w = tid >> 5, lane = tid & 31, l16 = lane & 15, hi = lane >> 4;
  const int wm = w >> 2, wn = w & 3;
#pragma unroll
  for (int mt = 0; mt < 4; ++mt) {
#pragma unroll
    for (int nt = 0; nt < 2; ++nt) {
      int c = n0 + wn * 32 + nt * 16 + l16;
      float bv = bias[c];
#pragma unroll
      for (int r = 0; r < 8; ++r) {
        int m = m0 + wm * 64 + mt * 16 + r + (hi ? 8 : 0);
        Out[(size_t)m * DIM + c] = acc[mt][nt][r] + bv;
      }
    }
  }
}

// ---------------------------------------------------------------------------
extern "C" void kernel_launch(void* const* d_in, const int* in_sizes, int n_in,
                              void* d_out, int out_size, void* d_ws,
                              size_t ws_size, hipStream_t stream) {
  const float* x      = (const float*)d_in[0];   // [8,1024,768]
  const float* w_qkv  = (const float*)d_in[1];   // [768,2304]
  const float* w_proj = (const float*)d_in[2];   // [768,768]
  const float* b_proj = (const float*)d_in[3];   // [768]
  float* out = (float*)d_out;

  const size_t X_n  = (size_t)MTOT * DIM;        // 6,291,456
  const size_t Wq_n = (size_t)DIM * 3 * DIM;     // 1,769,472
  const size_t Wp_n = (size_t)DIM * DIM;         //   589,824

  f16_t* ws    = (f16_t*)d_ws;                   // ~67.6 MB total f16 scratch
  f16_t* X16   = ws;  ws += X_n;
  f16_t* WqT16 = ws;  ws += Wq_n;                // [3*DIM][DIM] (transposed)
  f16_t* WpT16 = ws;  ws += Wp_n;                // [DIM][DIM]   (transposed)
  f16_t* Q16   = ws;  ws += X_n;                 // [B,H,N,D]
  f16_t* K16   = ws;  ws += X_n;                 // [B,H,N,D]
  f16_t* Vt16  = ws;  ws += X_n;                 // [B,H,D,N]
  f16_t* AO16  = ws;  ws += X_n;                 // [B,N,H*D]

  cvt_f32_f16<<<512, 256, 0, stream>>>(x, X16, (int)X_n);
  cvt_transpose_f32_f16<<<512, 256, 0, stream>>>(w_qkv, WqT16, DIM, 3 * DIM);
  cvt_transpose_f32_f16<<<256, 256, 0, stream>>>(w_proj, WpT16, DIM, DIM);

  gemm_qkv_kernel<<<dim3(MTOT / 128, (3 * DIM) / 128), 256, 0, stream>>>(
      X16, WqT16, Q16, K16, Vt16);

  attn_kernel<<<BATCH * NH * (SEQ / 64), 128, 0, stream>>>(
      Q16, K16, Vt16, AO16);

  gemm_proj_kernel<<<dim3(MTOT / 128, DIM / 128), 256, 0, stream>>>(
      AO16, WpT16, b_proj, out);
}
